// GCN_48369921688195
// MI455X (gfx1250) — compile-verified, matrix operand order checked by reading, not run on hardware
//
#include <hip/hip_runtime.h>
#include <hip/hip_bf16.h>

typedef __attribute__((ext_vector_type(16))) __bf16 v16bf;
typedef __attribute__((ext_vector_type(8)))  float  v8f;

#define HID 128
#define NGRAPH 256

// ---------------------------------------------------------------------------
// Degree / norm kernels
// ---------------------------------------------------------------------------
__global__ void gcn_deg_init(float* __restrict__ deg, int n) {
    int i = blockIdx.x * blockDim.x + threadIdx.x;
    if (i < n) deg[i] = 1.0f;                 // self-loop contributes 1
}

__global__ void gcn_deg_edges(const int* __restrict__ dst, float* __restrict__ deg, int e) {
    int i = blockIdx.x * blockDim.x + threadIdx.x;
    if (i < e) atomicAdd(&deg[dst[i]], 1.0f);
}

__global__ void gcn_deg_rsqrt(float* __restrict__ deg, int n) {
    int i = blockIdx.x * blockDim.x + threadIdx.x;
    if (i < n) deg[i] = rsqrtf(deg[i]);       // in-place -> dinv
}

// ---------------------------------------------------------------------------
// Fused WMMA GEMM + self-loop epilogue:
//   H[n,:]   = act(X[n,:]) @ W
//   AGG[n,:] = bias + dinv[n]^2 * H[n,:]     (edge scatter adds on top)
// AGG must NOT alias X (other blocks still read X while we store AGG).
// Block: 128 threads = 4 waves; each wave owns a 16-row tile -> 64 rows/block.
// ---------------------------------------------------------------------------
__global__ __launch_bounds__(128)
void gcn_gemm_wmma(const float* __restrict__ X, const float* __restrict__ W,
                   const float* __restrict__ dinv, const float* __restrict__ bias,
                   float* __restrict__ H, float* __restrict__ AGG,
                   int n_nodes, int relu_in)
{
    // sWB[((nt*4 + c)*32 + lane)*16 + i]
    __shared__ alignas(32) __bf16 sWB[8 * 4 * 32 * 16];   // 32 KB

    // cooperative pre-swizzle of W into B-operand layout
    for (int idx = threadIdx.x; idx < 8 * 4 * 32 * 16; idx += blockDim.x) {
        int i  = idx & 15;
        int l  = (idx >> 4) & 31;
        int c  = (idx >> 9) & 3;
        int nt = (idx >> 11) & 7;
        int kb = (l >> 4) * 8;
        int k  = 32 * c + ((i < 8) ? (kb + i) : (16 + kb + (i - 8)));
        int col = nt * 16 + (l & 15);
        sWB[idx] = (__bf16)W[k * HID + col];
    }
    __syncthreads();

    const int wave    = threadIdx.x >> 5;
    const int lane    = threadIdx.x & 31;
    const int halfSel = lane >> 4;            // 0: lanes 0-15, 1: lanes 16-31
    const int lrow    = lane & 15;
    const int kb      = halfSel * 8;
    const int rowBase = (blockIdx.x * 4 + wave) * 16;
    const bool fullTile = (rowBase + 16 <= n_nodes);   // wave-uniform guard

    // ---- load A operand: 4 chunks of K=32, bf16, documented 16-bit A layout
    int arow = rowBase + lrow;
    if (arow >= n_nodes) arow = n_nodes - 1;  // clamp; stores are guarded
    const float* xr = X + (size_t)arow * HID;

    v16bf A[4];
#pragma unroll
    for (int c = 0; c < 4; ++c) {
        float lo[8], hi[8];
        const float4* p = (const float4*)(xr + 32 * c + kb);        // kb in {0,8}: 16B aligned
        const float4* q = (const float4*)(xr + 32 * c + 16 + kb);
        float4 p0 = p[0], p1 = p[1], q0 = q[0], q1 = q[1];
        lo[0]=p0.x; lo[1]=p0.y; lo[2]=p0.z; lo[3]=p0.w;
        lo[4]=p1.x; lo[5]=p1.y; lo[6]=p1.z; lo[7]=p1.w;
        hi[0]=q0.x; hi[1]=q0.y; hi[2]=q0.z; hi[3]=q0.w;
        hi[4]=q1.x; hi[5]=q1.y; hi[6]=q1.z; hi[7]=q1.w;
#pragma unroll
        for (int i = 0; i < 8; ++i) {
            float f0 = lo[i], f1 = hi[i];
            if (relu_in) {                        // llvm.maxnum -> single v_max_num_f32
                f0 = __builtin_fmaxf(f0, 0.0f);
                f1 = __builtin_fmaxf(f1, 0.0f);
            }
            A[c][i]     = (__bf16)f0;
            A[c][i + 8] = (__bf16)f1;
        }
    }

    // ---- per-row dinv^2 for the epilogue (rows this lane will store)
    float d2[8];
#pragma unroll
    for (int r = 0; r < 8; ++r) {
        int row = rowBase + r + 8 * halfSel;
        int cr  = (row < n_nodes) ? row : (n_nodes - 1);
        float d = dinv[cr];
        d2[r] = d * d;
    }

    // ---- 8 output column tiles, 4 K-steps each
#pragma unroll
    for (int nt = 0; nt < 8; ++nt) {
        v8f acc = {};
#pragma unroll
        for (int c = 0; c < 4; ++c) {
            v16bf B = *(const v16bf*)&sWB[((nt * 4 + c) * 32 + lane) * 16];
            acc = __builtin_amdgcn_wmma_f32_16x16x32_bf16(
                false, A[c], false, B, (short)0, acc, false, false);
        }
        // D layout: VGPR r holds row (r + 8*halfSel), col = nt*16 + lrow
        const int col  = nt * 16 + lrow;
        const float bc = bias[col];
        if (fullTile) {
#pragma unroll
            for (int r = 0; r < 8; ++r) {
                int row = rowBase + r + 8 * halfSel;
                size_t off = (size_t)row * HID + col;
                float v = acc[r];
                H[off]   = v;
                AGG[off] = bc + d2[r] * v;
            }
        } else {
#pragma unroll
            for (int r = 0; r < 8; ++r) {
                int row = rowBase + r + 8 * halfSel;
                if (row < n_nodes) {
                    size_t off = (size_t)row * HID + col;
                    float v = acc[r];
                    H[off]   = v;
                    AGG[off] = bc + d2[r] * v;
                }
            }
        }
    }
}

// ---------------------------------------------------------------------------
// Edge scatter:  AGG[dst] += dinv[src]*dinv[dst] * H[src]
// One wave per edge, 4 features per lane (coalesced float4 gather).
// ---------------------------------------------------------------------------
__global__ void gcn_agg_edges(const float* __restrict__ H, const float* __restrict__ dinv,
                              const int* __restrict__ src, const int* __restrict__ dst,
                              float* __restrict__ A, int n_edges)
{
    int gw   = (int)(((long)blockIdx.x * blockDim.x + threadIdx.x) >> 5); // wave per edge
    int lane = threadIdx.x & 31;
    if (gw >= n_edges) return;
    int s = src[gw], d = dst[gw];
    float w = dinv[s] * dinv[d];
    float4 hv = *(const float4*)(H + (size_t)s * HID + lane * 4);
    float* out = A + (size_t)d * HID + lane * 4;
    atomicAdd(out + 0, w * hv.x);
    atomicAdd(out + 1, w * hv.y);
    atomicAdd(out + 2, w * hv.z);
    atomicAdd(out + 3, w * hv.w);
}

// ---------------------------------------------------------------------------
// Pooling + head
// ---------------------------------------------------------------------------
__global__ void gcn_pool_init(float* __restrict__ pooled, float* __restrict__ cnt) {
    int i = blockIdx.x * blockDim.x + threadIdx.x;
    if (i < NGRAPH * HID) pooled[i] = 0.0f;
    if (i < NGRAPH) cnt[i] = 0.0f;
}

__global__ void gcn_pool_accum(const float* __restrict__ H, const int* __restrict__ batch,
                               float* __restrict__ pooled, float* __restrict__ cnt,
                               int n_nodes)
{
    int gw   = (int)(((long)blockIdx.x * blockDim.x + threadIdx.x) >> 5); // wave per node
    int lane = threadIdx.x & 31;
    if (gw >= n_nodes) return;
    int g = batch[gw];
    float4 hv = *(const float4*)(H + (size_t)gw * HID + lane * 4);
    float* out = pooled + (size_t)g * HID + lane * 4;
    atomicAdd(out + 0, hv.x);
    atomicAdd(out + 1, hv.y);
    atomicAdd(out + 2, hv.z);
    atomicAdd(out + 3, hv.w);
    if (lane == 0) atomicAdd(&cnt[g], 1.0f);
}

__global__ void gcn_head(const float* __restrict__ pooled, const float* __restrict__ cnt,
                         const float* __restrict__ Wl, const float* __restrict__ bl,
                         float* __restrict__ out)
{
    int t = blockIdx.x * blockDim.x + threadIdx.x;   // 512 threads: (g, c)
    if (t >= NGRAPH * 2) return;
    int g = t >> 1, c = t & 1;
    float inv = 1.0f / __builtin_fmaxf(cnt[g], 1.0f);
    float s = 0.0f;
    for (int k = 0; k < HID; ++k)
        s += pooled[g * HID + k] * Wl[k * 2 + c];
    out[g * 2 + c] = s * inv + bl[c];
}

// ---------------------------------------------------------------------------
// Launcher — ping-pong three buffers so fused GEMM never writes its own input
// ---------------------------------------------------------------------------
extern "C" void kernel_launch(void* const* d_in, const int* in_sizes, int n_in,
                              void* d_out, int out_size, void* d_ws, size_t ws_size,
                              hipStream_t stream)
{
    const float* x    = (const float*)d_in[0];
    const int*   ei   = (const int*)d_in[1];
    const int*   batch= (const int*)d_in[2];
    const float* W1   = (const float*)d_in[3];
    const float* b1   = (const float*)d_in[4];
    const float* W2   = (const float*)d_in[5];
    const float* b2   = (const float*)d_in[6];
    const float* W3   = (const float*)d_in[7];
    const float* b3   = (const float*)d_in[8];
    const float* Wl   = (const float*)d_in[9];
    const float* bl   = (const float*)d_in[10];
    float* out = (float*)d_out;

    const int n_nodes = in_sizes[0] / HID;
    const int n_edges = in_sizes[1] / 2;
    const int* src = ei;
    const int* dst = ei + n_edges;

    // workspace layout
    float* dinv   = (float*)d_ws;                                // n_nodes
    float* h      = (float*)((char*)d_ws + (1 << 19));           // n_nodes*128
    float* aggA   = h    + (size_t)n_nodes * HID;                // n_nodes*128
    float* aggB   = aggA + (size_t)n_nodes * HID;                // n_nodes*128
    float* pooled = aggB + (size_t)n_nodes * HID;                // 256*128
    float* cnt    = pooled + NGRAPH * HID;                       // 256

    // --- degree / normalization
    gcn_deg_init <<<(n_nodes + 255) / 256, 256, 0, stream>>>(dinv, n_nodes);
    gcn_deg_edges<<<(n_edges + 255) / 256, 256, 0, stream>>>(dst, dinv, n_edges);
    gcn_deg_rsqrt<<<(n_nodes + 255) / 256, 256, 0, stream>>>(dinv, n_nodes);

    const int  gemm_blocks = (n_nodes + 63) / 64;
    const long edge_blocks = (((long)n_edges * 32) + 255) / 256;
    const long node_blocks = (((long)n_nodes * 32) + 255) / 256;

    // --- layer 1: in = x, selfloop+scatter -> aggA
    gcn_gemm_wmma<<<gemm_blocks, 128, 0, stream>>>(x,    W1, dinv, b1, h, aggA, n_nodes, 0);
    gcn_agg_edges<<<edge_blocks, 256, 0, stream>>>(h, dinv, src, dst, aggA, n_edges);

    // --- layer 2: in = relu(aggA), out -> aggB
    gcn_gemm_wmma<<<gemm_blocks, 128, 0, stream>>>(aggA, W2, dinv, b2, h, aggB, n_nodes, 1);
    gcn_agg_edges<<<edge_blocks, 256, 0, stream>>>(h, dinv, src, dst, aggB, n_edges);

    // --- layer 3: in = relu(aggB), out -> aggA
    gcn_gemm_wmma<<<gemm_blocks, 128, 0, stream>>>(aggB, W3, dinv, b3, h, aggA, n_nodes, 1);
    gcn_agg_edges<<<edge_blocks, 256, 0, stream>>>(h, dinv, src, dst, aggA, n_edges);

    // --- pool + head
    gcn_pool_init <<<(NGRAPH * HID + 255) / 256, 256, 0, stream>>>(pooled, cnt);
    gcn_pool_accum<<<node_blocks, 256, 0, stream>>>(aggA, batch, pooled, cnt, n_nodes);
    gcn_head      <<<2, 256, 0, stream>>>(pooled, cnt, Wl, bl, out);
}
